// HybridThoughtAwareAttention_32169305047326
// MI455X (gfx1250) — matrix operand order repaired and checked by reading, not compiled, not run on hardware
//
#include <hip/hip_runtime.h>

// ---------------------------------------------------------------------------
// HybridThoughtAwareAttention for MI455X (gfx1250, wave32, WMMA bf16,
// async global->LDS tile staging with ASYNCcnt double buffering)
// ---------------------------------------------------------------------------

typedef unsigned short u16;
typedef unsigned int   u32;
typedef __bf16         bf16;
typedef bf16  v16bf __attribute__((ext_vector_type(16)));
typedef float v8f   __attribute__((ext_vector_type(8)));

union Frag {
    uint4 q[2];   // 32 bytes
    v16bf v;      // 16 x bf16
};

__device__ inline u16 f2bf(float f) {
    u32 u = __float_as_uint(f);
    u32 r = (u + 0x7FFFu + ((u >> 16) & 1u)) >> 16;   // round-to-nearest-even
    return (u16)r;
}
__device__ inline float bf2f(u16 h) {
    return __uint_as_float(((u32)h) << 16);
}

__device__ inline v8f wmma_bf16(const Frag& a, const Frag& b, v8f c) {
    // D = A(16x32 bf16) x B(32x16 bf16) + C(16x16 f32)
    return __builtin_amdgcn_wmma_f32_16x16x32_bf16(
        /*neg_a=*/false, a.v, /*neg_b=*/false, b.v,
        /*c_mod=*/(short)0, c, /*reuse_a=*/false, /*reuse_b=*/false);
}

// ---- CDNA5 async global->LDS copy (tracked by ASYNCcnt, no VGPR transit) ---
__device__ inline void async_ld16(u32 lds_off, const void* g) {
    asm volatile("global_load_async_to_lds_b128 %0, %1, off"
                 :: "v"(lds_off), "v"((unsigned long long)(uintptr_t)g)
                 : "memory");
}
template <int N>
__device__ inline void wait_asynccnt() {
    asm volatile("s_wait_asynccnt %0" :: "n"(N) : "memory");
}

// ---------------------------------------------------------------------------
// fp32 -> bf16 conversion
// ---------------------------------------------------------------------------
__global__ void k_f32_to_bf16(const float* __restrict__ in, u16* __restrict__ out, int n) {
    int i = blockIdx.x * blockDim.x + threadIdx.x;
    if (i < n) out[i] = f2bf(in[i]);
}

// ---------------------------------------------------------------------------
// Tiled bf16 WMMA GEMM:  C[m,n] = sum_k A[m,k] * W[n,k] + bias[n]
//   mode 0: fp32 out, row-major [M,N]
//   mode 1: bf16 out, head-interleaved [B,H,S,D]  (m=(b,s), n=(h,d))
//   mode 2: bf16 out, head + transposed [B,H,D,S]
// Block tile 128x128, 8 waves (4x2), wave tile 32x64, K-step 32.
// Tiles staged via async global->LDS copies, double-buffered on ASYNCcnt.
// ---------------------------------------------------------------------------
#define GT_M 128
#define GT_N 128
#define GT_K 32
#define LDSS 40   // padded row stride (ushorts); 40*2=80B, multiple of 16B

__global__ __launch_bounds__(256) void k_gemm_bf16(
    const u16* __restrict__ A, const u16* __restrict__ W,
    const float* __restrict__ bias, float* __restrict__ Cf, u16* __restrict__ Cb,
    int M, int N, int K, int mode, int S, int Dh) {
    __shared__ __align__(16) u16 As[2][GT_M][LDSS];
    __shared__ __align__(16) u16 Bs[2][GT_N][LDSS];

    const int tid  = threadIdx.x;
    const int lane = tid & 31;
    const int wave = tid >> 5;
    const int wy = wave >> 1;     // 0..3  (rows)
    const int wx = wave & 1;      // 0..1  (cols)
    const int m0 = blockIdx.y * GT_M;
    const int n0 = blockIdx.x * GT_N;
    const int r    = lane & 15;
    const int half = lane >> 4;

    v8f acc[2][4];
    for (int i = 0; i < 2; i++)
        for (int j = 0; j < 4; j++)
            acc[i][j] = (v8f){0.f,0.f,0.f,0.f,0.f,0.f,0.f,0.f};

    // each thread issues 4 async 16B copies per tile (4 per wave on ASYNCcnt)
    auto issue_tile = [&](int buf, int k0) {
        for (int t = 0; t < 2; t++) {
            int idx = tid + t * 256;        // 0..511 vectors-of-8
            int row = idx >> 2;
            int c8  = (idx & 3) << 3;
            async_ld16((u32)(uintptr_t)&As[buf][row][c8],
                       &A[(size_t)(m0 + row) * K + k0 + c8]);
            async_ld16((u32)(uintptr_t)&Bs[buf][row][c8],
                       &W[(size_t)(n0 + row) * K + k0 + c8]);
        }
    };

    const int nk = K / GT_K;
    issue_tile(0, 0);

    for (int i = 0; i < nk; i++) {
        const int cur = i & 1;
        if (i + 1 < nk) {
            issue_tile(cur ^ 1, (i + 1) * GT_K);  // prefetch next tile
            wait_asynccnt<4>();                   // tile i landed; next 4 in flight
        } else {
            wait_asynccnt<0>();
        }
        __syncthreads();

        // A fragments: 16x32, lanes 0-15 hold K 0-15, lanes 16-31 hold K 16-31
        Frag af[2];
        for (int ii = 0; ii < 2; ii++) {
            int row = wy * 32 + ii * 16 + r;
            int kb  = half * 16;
            af[ii].q[0] = *reinterpret_cast<const uint4*>(&As[cur][row][kb]);
            af[ii].q[1] = *reinterpret_cast<const uint4*>(&As[cur][row][kb + 8]);
        }
        // B fragments: 32x16 (Kdim x N); lane holds 16 contiguous k of column n
        Frag bfr[4];
        for (int j = 0; j < 4; j++) {
            int row = wx * 64 + j * 16 + r;
            int kb  = half * 16;
            bfr[j].q[0] = *reinterpret_cast<const uint4*>(&Bs[cur][row][kb]);
            bfr[j].q[1] = *reinterpret_cast<const uint4*>(&Bs[cur][row][kb + 8]);
        }
        for (int ii = 0; ii < 2; ii++)
            for (int j = 0; j < 4; j++)
                acc[ii][j] = wmma_bf16(af[ii], bfr[j], acc[ii][j]);
        __syncthreads();   // buffer `cur` free to be overwritten next iteration
    }

    // store: C layout — VGPR rr, lane: M = rr + 8*half, N = lane&15
    for (int i = 0; i < 2; i++) {
        for (int j = 0; j < 4; j++) {
            for (int rr = 0; rr < 8; rr++) {
                int m = m0 + wy * 32 + i * 16 + half * 8 + rr;
                int n = n0 + wx * 64 + j * 16 + r;
                float v = acc[i][j][rr] + bias[n];
                if (mode == 0) {
                    Cf[(size_t)m * N + n] = v;
                } else {
                    int b = m / S, s = m % S;
                    int h = n / Dh, d = n % Dh;
                    int H = N / Dh;
                    size_t o = (mode == 1)
                        ? (((size_t)(b * H + h) * S + s) * Dh + d)
                        : (((size_t)(b * H + h) * Dh + d) * S + s);
                    Cb[o] = f2bf(v);
                }
            }
        }
    }
}

// ---------------------------------------------------------------------------
// Flash-style self-attention. One wave handles a 16-query tile over all keys.
//   Q, K : bf16 [B,H,S,D]   Vt : bf16 [B,H,D,S]
//   Out  : fp32 [B,S,E] (merged heads)
// ---------------------------------------------------------------------------
#define AT_D 128
__global__ __launch_bounds__(128) void k_attn(
    const u16* __restrict__ Q, const u16* __restrict__ Km, const u16* __restrict__ Vt,
    float* __restrict__ Out, int B, int H, int S, float scale) {
    __shared__ __align__(16) u16 pstage[4][16 * 32];

    const int lane = threadIdx.x & 31;
    const int wave = threadIdx.x >> 5;
    const int r    = lane & 15;
    const int half = lane >> 4;

    const int ntq  = S / 16;
    const int tile = blockIdx.x * 4 + wave;
    const int qt   = tile % ntq;
    const int bh   = tile / ntq;
    const size_t head = (size_t)bh * S * AT_D;

    // load 4 Q A-fragments (D = 4 chunks of 32)
    Frag qf[4];
    {
        const u16* qbase = Q + head + (size_t)(qt * 16 + r) * AT_D;
        for (int c = 0; c < 4; c++) {
            int kb = c * 32 + half * 16;
            qf[c].q[0] = *reinterpret_cast<const uint4*>(qbase + kb);
            qf[c].q[1] = *reinterpret_cast<const uint4*>(qbase + kb + 8);
        }
    }

    float mrow[8], lrow[8];
    v8f accO[8];
    for (int i = 0; i < 8; i++) {
        mrow[i] = -3.0e38f;
        lrow[i] = 0.0f;
        accO[i] = (v8f){0.f,0.f,0.f,0.f,0.f,0.f,0.f,0.f};
    }

    u16* ps = pstage[wave];

    for (int j0 = 0; j0 < S; j0 += 32) {
        // ---- scores: two 16x16 tiles (keys j0..+15, j0+16..+31) ----
        v8f sc[2];
        sc[0] = (v8f){0.f,0.f,0.f,0.f,0.f,0.f,0.f,0.f};
        sc[1] = (v8f){0.f,0.f,0.f,0.f,0.f,0.f,0.f,0.f};
        for (int sub = 0; sub < 2; sub++) {
            const u16* kbase = Km + head + (size_t)(j0 + sub * 16 + r) * AT_D;
            for (int c = 0; c < 4; c++) {
                Frag kf;
                int kb = c * 32 + half * 16;
                kf.q[0] = *reinterpret_cast<const uint4*>(kbase + kb);
                kf.q[1] = *reinterpret_cast<const uint4*>(kbase + kb + 8);
                sc[sub] = wmma_bf16(qf[c], kf, sc[sub]);
            }
        }

        // ---- online softmax, per row (row M = rr + 8*half) ----
        for (int rr = 0; rr < 8; rr++) {
            float s0 = sc[0][rr] * scale;
            float s1 = sc[1][rr] * scale;
            float t = fmaxf(s0, s1);
            for (int mk = 1; mk < 16; mk <<= 1) t = fmaxf(t, __shfl_xor(t, mk, 32));
            float mnew = fmaxf(mrow[rr], t);
            float p0 = __expf(s0 - mnew);
            float p1 = __expf(s1 - mnew);
            float rs = p0 + p1;
            for (int mk = 1; mk < 16; mk <<= 1) rs += __shfl_xor(rs, mk, 32);
            float corr = __expf(mrow[rr] - mnew);
            lrow[rr] = lrow[rr] * corr + rs;
            mrow[rr] = mnew;
            for (int n = 0; n < 8; n++) accO[n][rr] *= corr;
            // stage P (16x32) in bf16 for the A-operand repack
            int Mrow = rr + 8 * half;
            ps[Mrow * 32 + r]      = f2bf(p0);
            ps[Mrow * 32 + 16 + r] = f2bf(p1);
        }

        // re-load P in the 16x32 A-fragment layout (same-wave LDS RAW;
        // compiler inserts s_wait_dscnt)
        Frag pf;
        {
            int kb = half * 16;
            pf.q[0] = *reinterpret_cast<const uint4*>(&ps[r * 32 + kb]);
            pf.q[1] = *reinterpret_cast<const uint4*>(&ps[r * 32 + kb + 8]);
        }

        // ---- accO += P(16x32) x V(32x128) ----
        for (int n = 0; n < 8; n++) {
            Frag vf;
            const u16* vbase = Vt + (size_t)bh * AT_D * S
                             + (size_t)(n * 16 + r) * S + j0 + half * 16;
            vf.q[0] = *reinterpret_cast<const uint4*>(vbase);
            vf.q[1] = *reinterpret_cast<const uint4*>(vbase + 8);
            accO[n] = wmma_bf16(pf, vf, accO[n]);
        }
    }

    // ---- finalize + write merged-head fp32 ----
    const int h = bh % H, b = bh / H;
    const int E = H * AT_D;
    for (int n = 0; n < 8; n++) {
        for (int rr = 0; rr < 8; rr++) {
            int s = qt * 16 + rr + 8 * half;
            int d = n * 16 + r;
            Out[(size_t)(b * S + s) * E + h * AT_D + d] = accO[n][rr] / lrow[rr];
        }
    }
}

// ---------------------------------------------------------------------------
// Slot projection (tiny): out[j,n] = bias[n] + sum_k slots[j,k] * W[n,k]
// ---------------------------------------------------------------------------
__global__ void k_slot_proj(const float* __restrict__ slots, const float* __restrict__ W,
                            const float* __restrict__ bias, float* __restrict__ out, int E) {
    int n = blockIdx.x * blockDim.x + threadIdx.x;   // 0 .. 8*E-1
    int slot = n / E, col = n % E;
    const float* srow = slots + (size_t)slot * E;
    const float* wrow = W + (size_t)col * E;
    float a = bias[col];
    for (int k = 0; k < E; k++) a += srow[k] * wrow[k];
    out[n] = a;
}

// ---------------------------------------------------------------------------
// Thought cross-attention (8 keys): Comb += softmax(tq . tk / sqrt(D)) @ tv
// ---------------------------------------------------------------------------
__global__ void k_thought(const u16* __restrict__ TQ, const float* __restrict__ TK,
                          const float* __restrict__ TV, float* __restrict__ Comb,
                          int B, int H, int S, int E, float scale) {
    int t = blockIdx.x * blockDim.x + threadIdx.x;   // B*S*H threads
    if (t >= B * S * H) return;
    int h = t % H;
    int sb = t / H;
    int s = sb % S, b = sb / S;
    const int D = E / H;
    const u16* qrow = TQ + ((size_t)(b * H + h) * S + s) * D;

    float sc[8];
    for (int j = 0; j < 8; j++) sc[j] = 0.f;
    for (int d = 0; d < D; d++) {
        float qv = bf2f(qrow[d]);
        for (int j = 0; j < 8; j++) sc[j] += qv * TK[(size_t)j * E + h * D + d];
    }
    float mx = sc[0] * scale;
    for (int j = 1; j < 8; j++) mx = fmaxf(mx, sc[j] * scale);
    float den = 0.f, p[8];
    for (int j = 0; j < 8; j++) { p[j] = __expf(sc[j] * scale - mx); den += p[j]; }
    float inv = 1.0f / den;
    for (int j = 0; j < 8; j++) p[j] *= inv;

    float* out = Comb + (size_t)(b * S + s) * E + h * D;
    for (int d = 0; d < D; d++) {
        float o = 0.f;
        for (int j = 0; j < 8; j++) o += p[j] * TV[(size_t)j * E + h * D + d];
        out[d] += o;
    }
}

// ---------------------------------------------------------------------------
// LayerNorm over last dim (E) -> bf16
// ---------------------------------------------------------------------------
__global__ __launch_bounds__(256) void k_layernorm(
    const float* __restrict__ X, const float* __restrict__ gamma,
    const float* __restrict__ beta, u16* __restrict__ Out, int E) {
    __shared__ float red[256];
    const int row = blockIdx.x;
    const float* x = X + (size_t)row * E;

    float s = 0.f, s2 = 0.f;
    for (int c = threadIdx.x; c < E; c += 256) {
        float v = x[c];
        s += v; s2 += v * v;
    }
    red[threadIdx.x] = s; __syncthreads();
    for (int off = 128; off > 0; off >>= 1) {
        if (threadIdx.x < off) red[threadIdx.x] += red[threadIdx.x + off];
        __syncthreads();
    }
    float mean = red[0] / (float)E;
    __syncthreads();
    red[threadIdx.x] = s2; __syncthreads();
    for (int off = 128; off > 0; off >>= 1) {
        if (threadIdx.x < off) red[threadIdx.x] += red[threadIdx.x + off];
        __syncthreads();
    }
    float var = red[0] / (float)E - mean * mean;
    float inv = rsqrtf(var + 1e-5f);
    for (int c = threadIdx.x; c < E; c += 256)
        Out[(size_t)row * E + c] = f2bf((x[c] - mean) * inv * gamma[c] + beta[c]);
}

// ---------------------------------------------------------------------------
// Launch
// ---------------------------------------------------------------------------
extern "C" void kernel_launch(void* const* d_in, const int* in_sizes, int n_in,
                              void* d_out, int out_size, void* d_ws, size_t ws_size,
                              hipStream_t stream) {
    const float* x     = (const float*)d_in[0];
    const float* slots = (const float*)d_in[1];
    const float* Wq  = (const float*)d_in[2];  const float* bq  = (const float*)d_in[3];
    const float* Wk  = (const float*)d_in[4];  const float* bk  = (const float*)d_in[5];
    const float* Wv  = (const float*)d_in[6];  const float* bv  = (const float*)d_in[7];
    const float* Wtq = (const float*)d_in[8];  const float* btq = (const float*)d_in[9];
    const float* Wtk = (const float*)d_in[10]; const float* btk = (const float*)d_in[11];
    const float* Wtv = (const float*)d_in[12]; const float* btv = (const float*)d_in[13];
    const float* Wo  = (const float*)d_in[14]; const float* bo  = (const float*)d_in[15];
    const float* gamma = (const float*)d_in[16];
    const float* beta  = (const float*)d_in[17];

    const int B = 2, S = 2048, E = 2048, H = 16, D = 128;
    const int M = B * S;                 // 4096
    const float scale = 0.08838834764831845f;  // 1/sqrt(128)

    // workspace carve-out
    char* ws = (char*)d_ws;
    size_t off = 0;
    auto carve = [&](size_t bytes) -> void* {
        void* p = ws + off;
        off += (bytes + 255) & ~(size_t)255;
        return p;
    };
    u16* xb   = (u16*)carve((size_t)M * E * 2);
    u16* wqb  = (u16*)carve((size_t)E * E * 2);
    u16* wkb  = (u16*)carve((size_t)E * E * 2);
    u16* wvb  = (u16*)carve((size_t)E * E * 2);
    u16* wtqb = (u16*)carve((size_t)E * E * 2);
    u16* wob  = (u16*)carve((size_t)E * E * 2);
    u16* qh   = (u16*)carve((size_t)M * E * 2);   // [B,H,S,D]
    u16* kh   = (u16*)carve((size_t)M * E * 2);   // [B,H,S,D]
    u16* vt   = (u16*)carve((size_t)M * E * 2);   // [B,H,D,S]
    u16* tqh  = (u16*)carve((size_t)M * E * 2);   // [B,H,S,D]
    float* tk = (float*)carve((size_t)8 * E * 4);
    float* tv = (float*)carve((size_t)8 * E * 4);
    float* comb   = (float*)carve((size_t)M * E * 4);
    u16*   normed = (u16*)carve((size_t)M * E * 2);

    // 1) fp32 -> bf16 conversions
    {
        int n = M * E;
        k_f32_to_bf16<<<(n + 255) / 256, 256, 0, stream>>>(x, xb, n);
        n = E * E;
        k_f32_to_bf16<<<(n + 255) / 256, 256, 0, stream>>>(Wq,  wqb,  n);
        k_f32_to_bf16<<<(n + 255) / 256, 256, 0, stream>>>(Wk,  wkb,  n);
        k_f32_to_bf16<<<(n + 255) / 256, 256, 0, stream>>>(Wv,  wvb,  n);
        k_f32_to_bf16<<<(n + 255) / 256, 256, 0, stream>>>(Wtq, wtqb, n);
        k_f32_to_bf16<<<(n + 255) / 256, 256, 0, stream>>>(Wo,  wob,  n);
    }

    // 2) projections via WMMA GEMM (async-LDS staged)
    dim3 ggrid(E / GT_N, M / GT_M);   // (16, 32)
    k_gemm_bf16<<<ggrid, 256, 0, stream>>>(xb, wqb,  bq,  nullptr, qh,  M, E, E, 1, S, D);
    k_gemm_bf16<<<ggrid, 256, 0, stream>>>(xb, wkb,  bk,  nullptr, kh,  M, E, E, 1, S, D);
    k_gemm_bf16<<<ggrid, 256, 0, stream>>>(xb, wvb,  bv,  nullptr, vt,  M, E, E, 2, S, D);
    k_gemm_bf16<<<ggrid, 256, 0, stream>>>(xb, wtqb, btq, nullptr, tqh, M, E, E, 1, S, D);

    // 3) slot projections (tiny, fp32)
    k_slot_proj<<<(8 * E) / 256, 256, 0, stream>>>(slots, Wtk, btk, tk, E);
    k_slot_proj<<<(8 * E) / 256, 256, 0, stream>>>(slots, Wtv, btv, tv, E);

    // 4) flash self-attention -> comb (fp32, merged heads)
    {
        int tiles = B * H * (S / 16);       // 4096
        k_attn<<<tiles / 4, 128, 0, stream>>>(qh, kh, vt, comb, B, H, S, scale);
    }

    // 5) thought cross-attention, accumulated into comb
    {
        int n = B * S * H;                  // 65536
        k_thought<<<(n + 255) / 256, 256, 0, stream>>>(tqh, tk, tv, comb, B, H, S, E, scale);
    }

    // 6) layernorm -> bf16
    k_layernorm<<<M, 256, 0, stream>>>(comb, gamma, beta, normed, E);

    // 7) output projection -> d_out (fp32)
    k_gemm_bf16<<<ggrid, 256, 0, stream>>>(normed, wob, bo, (float*)d_out, nullptr,
                                           M, E, E, 0, S, D);
}
